// FM_LOSS_12146167513244
// MI455X (gfx1250) — matrix operand order repaired and checked by reading, not compiled, not run on hardware
//
#include <hip/hip_runtime.h>
#include <hip/hip_bf16.h>

typedef __attribute__((ext_vector_type(16))) __bf16 v16bf;
typedef __attribute__((ext_vector_type(8)))  float  v8f;
typedef __attribute__((ext_vector_type(2)))  __bf16 bf16x2;

#define B_  8
#define C_  512
#define N_  2048
#define H_  8
#define D_  64
#define BN_EPS 1e-5f

#define QSTR 72   // padded d-stride (72 elems = 144 B, 16B-aligned rows)
#define TSTR 72
#define WSTR 40   // padded c-stride for 32-wide embed tiles

union FragBF { v16bf v; bf16x2 p[8]; };

static __device__ inline unsigned short bfbits(float f) {
  // round-to-nearest-even f32 -> bf16 bit pattern
  unsigned u = __builtin_bit_cast(unsigned, f);
  return (unsigned short)((u + 0x7FFFu + ((u >> 16) & 1u)) >> 16);
}
static __device__ inline __bf16 b16(float f) {
  unsigned short s = bfbits(f);
  return __builtin_bit_cast(__bf16, s);
}
static __device__ inline v8f zero8() {
  v8f z = {0.f,0.f,0.f,0.f,0.f,0.f,0.f,0.f};
  return z;
}

// gfx1250 async copy: global -> LDS, 16B per lane, tracked by ASYNCcnt
static __device__ inline void async_ld16(unsigned lds_off, const void* g) {
  asm volatile("global_load_async_to_lds_b128 %0, %1, off"
               :: "v"(lds_off), "v"((unsigned long long)(size_t)g)
               : "memory");
}
static __device__ inline void wait_async0() {
  asm volatile("s_wait_asynccnt 0" ::: "memory");
}

// ---------------------------------------------------------------------------
// Kernel 0: one-time bf16 conversion of f_s into two layouts:
//   fsD [b][h][d][n]  (same layout as f_s, bf16)          -> V-as-B tiles
//   fsT [b][h][n][d]  (transposed within each head)       -> Q rows / K-as-B tiles
// ---------------------------------------------------------------------------
__global__ __launch_bounds__(256) void cvt_transpose_kernel(
    const float* __restrict__ f_s, unsigned short* __restrict__ fsT,
    unsigned short* __restrict__ fsD)
{
  __shared__ unsigned short Tl[64 * 65];
  const int tid = threadIdx.x;
  const int n0  = blockIdx.x * 64;
  const int h   = blockIdx.y;
  const int b   = blockIdx.z;
  const float* src = f_s + ((size_t)b * C_ + (size_t)h * D_) * N_;
  unsigned short* dD = fsD + ((size_t)b * C_ + (size_t)h * D_) * N_;
  unsigned short* dT = fsT + ((size_t)(b * H_ + h) * N_) * D_;

  for (int i = tid; i < 64 * 64; i += 256) {
    int d = i >> 6, n = i & 63;               // coalesced over n
    unsigned short v = bfbits(src[(size_t)d * N_ + n0 + n]);
    dD[(size_t)d * N_ + n0 + n] = v;
    Tl[n * 65 + d] = v;
  }
  __syncthreads();
  for (int i = tid; i < 64 * 64; i += 256) {
    int n = i >> 6, d = i & 63;               // coalesced over d
    dT[((size_t)(n0 + n)) * D_ + d] = Tl[n * 65 + d];
  }
}

// ---------------------------------------------------------------------------
// Kernel 1: X[b,o,j] = sum_c W[o,c] * f_t[b,c,j]   (Conv1d k=1, bias-free)
// ---------------------------------------------------------------------------
__global__ __launch_bounds__(256) void embed_gemm_kernel(
    const float* __restrict__ W, const float* __restrict__ f_t,
    float* __restrict__ Xbuf)
{
  __shared__ __align__(16) __bf16 Ws[128 * WSTR]; // [o_local][c_local]
  __shared__ __align__(16) __bf16 Ft[64 * WSTR];  // [j_local][c_local]

  const int tid  = threadIdx.x;
  const int wave = tid >> 5;
  const int lane = tid & 31;
  const int ln   = lane & 15;
  const int hh   = lane >> 4;

  const int j0 = blockIdx.x * 64;
  const int o0 = blockIdx.y * 128;
  const int b  = blockIdx.z;

  v8f acc[4];
  #pragma unroll
  for (int s = 0; s < 4; ++s) acc[s] = zero8();

  for (int c0 = 0; c0 < C_; c0 += 32) {
    __syncthreads();
    for (int i = tid; i < 128 * 32; i += 256) {
      int o = i >> 5, cc = i & 31;
      Ws[o * WSTR + cc] = b16(W[(size_t)(o0 + o) * C_ + c0 + cc]);
    }
    for (int i = tid; i < 64 * 32; i += 256) {
      int cc = i >> 6, j = i & 63;
      Ft[j * WSTR + cc] = b16(f_t[((size_t)b * C_ + c0 + cc) * N_ + j0 + j]);
    }
    __syncthreads();

    FragBF wa;
    #pragma unroll
    for (int vv = 0; vv < 8; ++vv) {
      int c = (vv >> 2) * 16 + hh * 8 + (vv & 3) * 2;
      wa.p[vv] = *(const bf16x2*)&Ws[(wave * 16 + ln) * WSTR + c];
    }
    #pragma unroll
    for (int s = 0; s < 4; ++s) {
      FragBF fb;
      #pragma unroll
      for (int vv = 0; vv < 8; ++vv) {
        int c = (vv >> 2) * 16 + hh * 8 + (vv & 3) * 2;
        fb.p[vv] = *(const bf16x2*)&Ft[(s * 16 + ln) * WSTR + c];
      }
      acc[s] = __builtin_amdgcn_wmma_f32_16x16x32_bf16(
          false, wa.v, false, fb.v, (short)0, acc[s], false, false);
    }
  }

  #pragma unroll
  for (int s = 0; s < 4; ++s)
    #pragma unroll
    for (int v = 0; v < 8; ++v) {
      int ol = wave * 16 + hh * 8 + v;
      int j  = j0 + s * 16 + ln;
      Xbuf[((size_t)b * C_ + o0 + ol) * N_ + j] = acc[s][v];
    }
}

// ---------------------------------------------------------------------------
// Kernel 2: per-channel mean / rstd over (b, n), deterministic tree reduction
// ---------------------------------------------------------------------------
__global__ __launch_bounds__(256) void bn_stats_kernel(
    const float* __restrict__ X, float2* __restrict__ mv)
{
  __shared__ float s1[256];
  __shared__ float s2[256];
  const int ch = blockIdx.x;
  const int tid = threadIdx.x;
  float a = 0.f, q = 0.f;
  for (int i = tid; i < B_ * N_; i += 256) {
    int b = i >> 11, j = i & (N_ - 1);
    float x = X[((size_t)b * C_ + ch) * N_ + j];
    a += x; q += x * x;
  }
  s1[tid] = a; s2[tid] = q;
  __syncthreads();
  for (int s = 128; s > 0; s >>= 1) {
    if (tid < s) { s1[tid] += s1[tid + s]; s2[tid] += s2[tid + s]; }
    __syncthreads();
  }
  if (tid == 0) {
    const float inv = 1.f / (float)(B_ * N_);
    float mean = s1[0] * inv;
    float var  = s2[0] * inv - mean * mean;   // biased, matches jnp.var
    mv[ch] = make_float2(mean, rsqrtf(var + BN_EPS));
  }
}

// ---------------------------------------------------------------------------
// Kernel 3: flash attention (q=k=v) fused with MSE vs BN(embed).
// WG = 128 query rows of one (b,h); 8 waves x 16 rows.
// K/V tiles stream via GLOBAL_LOAD_ASYNC_TO_LDS_B128, double-buffered.
// ---------------------------------------------------------------------------
__global__ __launch_bounds__(256) void attn_loss_kernel(
    const unsigned short* __restrict__ fsT, const unsigned short* __restrict__ fsD,
    const float* __restrict__ Xbuf, const float2* __restrict__ mv,
    const float* __restrict__ gamma, const float* __restrict__ beta,
    float* __restrict__ partials)
{
  __shared__ __align__(16) unsigned short Qs[128 * QSTR];     // [row][d]
  __shared__ __align__(16) unsigned short Ks[2 * 64 * QSTR];  // [buf][n'][d]
  __shared__ __align__(16) unsigned short Vt[2 * 64 * TSTR];  // [buf][d][n']
  __shared__ __align__(16) unsigned short Ps[8 * 16 * TSTR];  // per-wave P relayout
  __shared__ float red[256];

  const int tid  = threadIdx.x;
  const int wave = tid >> 5;
  const int lane = tid & 31;
  const int ln   = lane & 15;
  const int hh   = lane >> 4;

  const int row0 = blockIdx.x * 128;
  const int h    = blockIdx.y;
  const int b    = blockIdx.z;

  const unsigned short* fst = fsT + ((size_t)(b * H_ + h) * N_) * D_; // [n][d]
  const unsigned short* fsd = fsD + ((size_t)b * C_ + (size_t)h * D_) * N_; // [d][n]

  const unsigned qsBase = (unsigned)(size_t)&Qs[0];
  const unsigned ksBase = (unsigned)(size_t)&Ks[0];
  const unsigned vtBase = (unsigned)(size_t)&Vt[0];

  // ---- async: Q tile [128 rows x 64 d], 16B chunks from fsT ----
  #pragma unroll
  for (int k = 0; k < 4; ++k) {
    int idx = tid + k * 256;                 // 0..1023
    int j = idx >> 3, c8 = idx & 7;
    async_ld16(qsBase + (unsigned)(j * QSTR + c8 * 8) * 2,
               fst + ((size_t)(row0 + j)) * D_ + c8 * 8);
  }

  // issue K/V tile t into buffer sel
  auto issue_kv = [&](int t, int sel) {
    const int n0 = t * 64;
    const unsigned kb = ksBase + (unsigned)(sel * 64 * QSTR) * 2;
    const unsigned vb = vtBase + (unsigned)(sel * 64 * TSTR) * 2;
    #pragma unroll
    for (int k = 0; k < 2; ++k) {
      int idx = tid + k * 256;               // 0..511
      int r = idx >> 3, c8 = idx & 7;
      async_ld16(kb + (unsigned)(r * QSTR + c8 * 8) * 2,
                 fst + ((size_t)(n0 + r)) * D_ + c8 * 8);
      async_ld16(vb + (unsigned)(r * TSTR + c8 * 8) * 2,
                 fsd + (size_t)r * N_ + n0 + c8 * 8);
    }
  };

  const int NT = N_ / 64;
  issue_kv(0, 0);
  wait_async0();
  __syncthreads();        // Q + tile0 resident in LDS for all waves

  // ---- Q A-fragments (two K=32 halves of d) ----
  FragBF qa[2];
  {
    const unsigned short* qrow = &Qs[(wave * 16 + ln) * QSTR];
    #pragma unroll
    for (int kh = 0; kh < 2; ++kh)
      #pragma unroll
      for (int vv = 0; vv < 8; ++vv) {
        int d = kh * 32 + (vv >> 2) * 16 + hh * 8 + (vv & 3) * 2;
        qa[kh].p[vv] = *(const bf16x2*)&qrow[d];
      }
  }
  if (NT > 1) issue_kv(1, 1);   // prefetch tile 1 while tile 0 computes

  v8f oacc[4];
  #pragma unroll
  for (int s = 0; s < 4; ++s) oacc[s] = zero8();
  float mrow[8], lrow[8];
  #pragma unroll
  for (int v = 0; v < 8; ++v) { mrow[v] = -3.0e38f; lrow[v] = 0.f; }

  unsigned short* Pw = &Ps[wave * 16 * TSTR];

  for (int t = 0; t < NT; ++t) {
    const unsigned short* Kc = &Ks[(t & 1) * 64 * QSTR];
    const unsigned short* Vc = &Vt[(t & 1) * 64 * TSTR];

    // ---- S = Q K^T : 4 column subtiles of 16, K-depth 64 = 2 WMMAs ----
    v8f sacc[4];
    #pragma unroll
    for (int s = 0; s < 4; ++s) {
      sacc[s] = zero8();
      #pragma unroll
      for (int kh = 0; kh < 2; ++kh) {
        FragBF kb;
        #pragma unroll
        for (int vv = 0; vv < 8; ++vv) {
          int d = kh * 32 + (vv >> 2) * 16 + hh * 8 + (vv & 3) * 2;
          kb.p[vv] = *(const bf16x2*)&Kc[(s * 16 + ln) * QSTR + d];
        }
        sacc[s] = __builtin_amdgcn_wmma_f32_16x16x32_bf16(
            false, qa[kh].v, false, kb.v, (short)0, sacc[s], false, false);
      }
    }

    // ---- online softmax ----
    float mx[8];
    #pragma unroll
    for (int v = 0; v < 8; ++v) {
      float m0 = fmaxf(fmaxf(sacc[0][v], sacc[1][v]), fmaxf(sacc[2][v], sacc[3][v]));
      #pragma unroll
      for (int off = 1; off < 16; off <<= 1)
        m0 = fmaxf(m0, __shfl_xor(m0, off, 32));
      mx[v] = m0;
    }
    float scale[8];
    #pragma unroll
    for (int v = 0; v < 8; ++v) {
      float mn = fmaxf(mrow[v], mx[v]);
      scale[v] = __expf(mrow[v] - mn);
      mrow[v]  = mn;
      lrow[v] *= scale[v];
    }
    #pragma unroll
    for (int s = 0; s < 4; ++s)
      #pragma unroll
      for (int v = 0; v < 8; ++v)
        oacc[s][v] *= scale[v];

    float rs[8];
    #pragma unroll
    for (int v = 0; v < 8; ++v) rs[v] = 0.f;
    #pragma unroll
    for (int s = 0; s < 4; ++s)
      #pragma unroll
      for (int v = 0; v < 8; ++v) {
        float p = __expf(sacc[s][v] - mrow[v]);
        rs[v] += p;
        Pw[(hh * 8 + v) * TSTR + s * 16 + ln] = bfbits(p); // C/D -> [row][n']
      }
    #pragma unroll
    for (int v = 0; v < 8; ++v) {
      float r = rs[v];
      #pragma unroll
      for (int off = 1; off < 16; off <<= 1)
        r += __shfl_xor(r, off, 32);
      lrow[v] += r;
    }

    // DS ops are wave-in-order; fence the P relayout round-trip explicitly
    asm volatile("s_wait_dscnt 0" ::: "memory");

    // ---- O += P V : K-depth 64(n') = 2 halves, 4 dd subtiles ----
    #pragma unroll
    for (int nh = 0; nh < 2; ++nh) {
      FragBF pa;
      #pragma unroll
      for (int vv = 0; vv < 8; ++vv) {
        int np = nh * 32 + (vv >> 2) * 16 + hh * 8 + (vv & 3) * 2;
        pa.p[vv] = *(const bf16x2*)&Pw[ln * TSTR + np];
      }
      #pragma unroll
      for (int s = 0; s < 4; ++s) {
        FragBF vbf;
        #pragma unroll
        for (int vv = 0; vv < 8; ++vv) {
          int np = nh * 32 + (vv >> 2) * 16 + hh * 8 + (vv & 3) * 2;
          vbf.p[vv] = *(const bf16x2*)&Vc[(s * 16 + ln) * TSTR + np];
        }
        oacc[s] = __builtin_amdgcn_wmma_f32_16x16x32_bf16(
            false, pa.v, false, vbf.v, (short)0, oacc[s], false, false);
      }
    }

    // ---- rotate buffers: wait for t+1 data, then prefetch t+2 ----
    if (t + 1 < NT) {
      wait_async0();        // tile t+1 (issued earlier) now resident
      __syncthreads();      // all waves done reading buffer (t&1)
      if (t + 2 < NT) issue_kv(t + 2, t & 1);
    }
  }

  // ---- epilogue: O /= l, fuse MSE against BN-normalized embed target ----
  float lacc = 0.f;
  #pragma unroll
  for (int v = 0; v < 8; ++v) {
    float inv = 1.f / lrow[v];
    int row = row0 + wave * 16 + hh * 8 + v;
    #pragma unroll
    for (int s = 0; s < 4; ++s) {
      int dd = s * 16 + ln;
      int ch = h * D_ + dd;
      float o   = oacc[s][v] * inv;
      float x   = Xbuf[((size_t)b * C_ + ch) * N_ + row];
      float2 ms = mv[ch];
      float tgt = (x - ms.x) * ms.y * gamma[ch] + beta[ch];
      float dlt = o - tgt;
      lacc += dlt * dlt;
    }
  }
  red[tid] = lacc;
  __syncthreads();
  for (int s = 128; s > 0; s >>= 1) {
    if (tid < s) red[tid] += red[tid + s];
    __syncthreads();
  }
  if (tid == 0) {
    int flat = (blockIdx.z * gridDim.y + blockIdx.y) * gridDim.x + blockIdx.x;
    partials[flat] = red[0];
  }
}

// ---------------------------------------------------------------------------
// Kernel 4: deterministic final reduction -> scalar loss
// ---------------------------------------------------------------------------
__global__ __launch_bounds__(256) void finalize_kernel(
    const float* __restrict__ partials, int n, float* __restrict__ out)
{
  __shared__ float red[256];
  float a = 0.f;
  for (int i = threadIdx.x; i < n; i += 256) a += partials[i];
  red[threadIdx.x] = a;
  __syncthreads();
  for (int s = 128; s > 0; s >>= 1) {
    if (threadIdx.x < s) red[threadIdx.x] += red[threadIdx.x + s];
    __syncthreads();
  }
  if (threadIdx.x == 0)
    out[0] = red[0] * (1.0f / ((float)B_ * (float)C_ * (float)N_));
}

extern "C" void kernel_launch(void* const* d_in, const int* in_sizes, int n_in,
                              void* d_out, int out_size, void* d_ws, size_t ws_size,
                              hipStream_t stream) {
  const float* f_s   = (const float*)d_in[0];
  const float* f_t   = (const float*)d_in[1];
  const float* W     = (const float*)d_in[2];
  const float* gamma = (const float*)d_in[3];
  const float* beta  = (const float*)d_in[4];
  float* out = (float*)d_out;

  char* ws = (char*)d_ws;
  const size_t XB  = (size_t)B_ * C_ * N_ * sizeof(float);          // 33.5 MB
  const size_t FB  = (size_t)B_ * C_ * N_ * sizeof(unsigned short); // 16.8 MB
  float*          Xbuf     = (float*)ws;
  unsigned short* fsT      = (unsigned short*)(ws + XB);
  unsigned short* fsD      = (unsigned short*)(ws + XB + FB);
  float2*         mv       = (float2*)(ws + XB + 2 * FB);
  float*          partials = (float*)(ws + XB + 2 * FB + 4096);

  cvt_transpose_kernel<<<dim3(N_ / 64, H_, B_), 256, 0, stream>>>(f_s, fsT, fsD);
  embed_gemm_kernel<<<dim3(N_ / 64, C_ / 128, B_), 256, 0, stream>>>(W, f_t, Xbuf);
  bn_stats_kernel<<<C_, 256, 0, stream>>>(Xbuf, mv);
  attn_loss_kernel<<<dim3(N_ / 128, H_, B_), 256, 0, stream>>>(
      fsT, fsD, Xbuf, mv, gamma, beta, partials);
  finalize_kernel<<<1, 256, 0, stream>>>(partials, (N_ / 128) * H_ * B_, out);
}
